// HaloWindowAttention_69818988363885
// MI455X (gfx1250) — matrix operand order repaired and checked by reading, not compiled, not run on hardware
//
#include <hip/hip_runtime.h>
#include <hip/hip_bf16.h>
#include <math.h>

// ---------------------------------------------------------------------------
// HaloWindowAttention on MI455X (gfx1250): bf16 WMMA pipeline.
// B=512, C=128, H=W=16, WS=16, HALO=4, HEADS=4, HD=32, WSH=24, NTOK=576.
// ---------------------------------------------------------------------------

#define CC     128
#define NTOK   576          // 24*24 tokens per (padded) window
#define CSZ    256          // center query tokens
#define CSTART 100          // HALO*WSH + HALO
#define HEADS  4
#define HD     32
#define SCALE  0.17677669529663687f   // 32^-0.5

typedef __attribute__((ext_vector_type(16))) __bf16 v16bf;
typedef __attribute__((ext_vector_type(8)))  __bf16 v8bf;
typedef __attribute__((ext_vector_type(4)))  __bf16 v4bf;
typedef __attribute__((ext_vector_type(8)))  float  v8f;

// ---- WMMA fragment loaders (wave32 layouts, CDNA5 ISA 7.12.2) ---

// A fragment (16x32 bf16, MxK) from row-major [M][K], row stride = `stride`.
__device__ __forceinline__ v16bf load_a16(const __bf16* p, int stride) {
  const int lane = threadIdx.x & 31;
  const int r    = lane & 15;
  const int kh   = (lane >> 4) << 4;   // 0 or 16
  const __bf16* q = p + r * stride + kh;
  v16bf a;
#pragma unroll
  for (int j = 0; j < 16; ++j) a[j] = q[j];
  return a;
}

// B fragment (32x16 bf16, KxN) from row-major [N][K] storage (B^T, e.g. W[n][k]).
__device__ __forceinline__ v16bf load_bt16(const __bf16* p, int stride) {
  const int lane = threadIdx.x & 31;
  const int n    = lane & 15;
  const int kh   = (lane >> 4) << 4;   // 0 or 16
  const __bf16* q = p + n * stride + kh;
  v16bf b;
#pragma unroll
  for (int j = 0; j < 16; ++j) b[j] = q[j];
  return b;
}

__device__ __forceinline__ v8f wmma_bf16(v16bf a, v16bf b, v8f c) {
  return __builtin_amdgcn_wmma_f32_16x16x32_bf16(false, a, false, b,
                                                 (short)0, c, false, false);
}

// Async global->LDS bulk copy (gfx1250 GLOBAL_LOAD_ASYNC_TO_LDS_B128, GVS mode:
// mem_addr = SGPR64 + VGPR32 offset; VDST VGPR = LDS byte address).  Tracked by
// ASYNCcnt; caller must s_wait_asynccnt + barrier before reading LDS.
__device__ __forceinline__ void async_copy16(const void* gsrc, void* ldst,
                                             int nbytes, int tid) {
  const unsigned lbase = (unsigned)(uintptr_t)ldst;            // LDS offset in low 32b
  const unsigned long long gb = (unsigned long long)(uintptr_t)gsrc;
  for (int off = tid * 16; off < nbytes; off += 256 * 16) {
    asm volatile("global_load_async_to_lds_b128 %0, %1, %2"
                 :: "v"(lbase + (unsigned)off), "v"(off), "s"(gb)
                 : "memory");
  }
}

__device__ __forceinline__ void wait_async_all() {
  asm volatile("s_wait_asynccnt 0" ::: "memory");
}

// ---------------------------------------------------------------------------
// Kernel 1: reflect-pad gather + QKV projection.  One workgroup per window.
//   X_lds [576][128] bf16 + W_lds [384][128] bf16 + bias (~242 KB dynamic LDS).
//   Output: qc [w][head][256][32], k [w][head][576][32],
//           v TRANSPOSED [w][head][32][576]        (all bf16)
// ---------------------------------------------------------------------------
__global__ __launch_bounds__(256) void halo_qkv_kernel(
    const float* __restrict__ x, const float* __restrict__ qkv_w,
    const float* __restrict__ qkv_b,
    __bf16* __restrict__ qc, __bf16* __restrict__ kk, __bf16* __restrict__ vv) {
  extern __shared__ char smem[];
  __bf16* Xl   = (__bf16*)smem;                   // 576*128
  __bf16* Wl   = Xl + NTOK * CC;                  // 384*128
  float*  bias = (float*)(Wl + 3 * CC * CC);      // 384

  const int w   = blockIdx.x;
  const int tid = threadIdx.x;
  const float* xw = x + (size_t)w * CC * 256;

  __builtin_prefetch(xw + tid * 64, 0, 0);        // global_prefetch_b8

  // gather X with reflect padding, f32 -> bf16
  for (int idx = tid; idx < NTOK * CC; idx += 256) {
    int ch = idx / NTOK, t = idx - ch * NTOK;
    int py = t / 24, px = t - py * 24;
    int y = py - 4; y = (y < 0) ? -y : ((y > 15) ? 30 - y : y);
    int z = px - 4; z = (z < 0) ? -z : ((z > 15) ? 30 - z : z);
    Xl[t * CC + ch] = (__bf16)xw[ch * 256 + y * 16 + z];
  }
  // weights f32 -> bf16, vectorized
  for (int i = tid * 4; i < 3 * CC * CC; i += 1024) {
    float4 f = *(const float4*)&qkv_w[i];
    v4bf b; b[0] = (__bf16)f.x; b[1] = (__bf16)f.y; b[2] = (__bf16)f.z; b[3] = (__bf16)f.w;
    *(v4bf*)&Wl[i] = b;
  }
  for (int idx = tid; idx < 3 * CC; idx += 256) bias[idx] = qkv_b[idx];
  __syncthreads();

  const int wv = tid >> 5, lane = tid & 31;
  const int l15 = lane & 15;
  const int mhalf = (lane >> 4) << 3;

  // N-tiles are 16 wide -> each lies entirely in one of q/k/v: branch is
  // tile-uniform (scalar).  3 N-tiles per wave (24 total), 36 M-tiles.
  for (int s = 0; s < 3; ++s) {
    const int nt    = wv + s * 8;         // 0..23
    const int g     = nt >> 3;            // 0:q 1:k 2:v   (uniform)
    const int head  = (nt >> 1) & 3;      // uniform
    const int dbase = (nt & 1) << 4;      // uniform
    const int d     = dbase + l15;
    const float bn  = bias[nt * 16 + l15];

    const long long qbase = (long long)(((size_t)w * HEADS + head) * CSZ) * HD
                            + d - (long long)CSTART * HD;
    const long long kbase = (long long)(((size_t)w * HEADS + head) * NTOK) * HD + d;
    const long long vbase = ((long long)((size_t)w * HEADS + head) * HD + d) * NTOK;

    for (int mt = 0; mt < 36; ++mt) {
      // q-tiles fully outside center slice [100,356): skip compute entirely
      if (g == 0 && (mt < 6 || mt > 22)) continue;
      v8f acc = {};
#pragma unroll
      for (int k4 = 0; k4 < 4; ++k4) {
        v16bf a = load_a16(&Xl[(mt * 16) * CC + k4 * 32], CC);
        v16bf b = load_bt16(&Wl[(nt * 16) * CC + k4 * 32], CC);
        acc = wmma_bf16(a, b, acc);
      }
      const int mbase = mt * 16 + mhalf;
      if (g == 2) {
        // transposed, contiguous along m: one 16B store per lane
        v8bf pk;
#pragma unroll
        for (int i = 0; i < 8; ++i) pk[i] = (__bf16)(acc[i] + bn);
        *(v8bf*)&vv[vbase + mbase] = pk;
      } else if (g == 1) {
#pragma unroll
        for (int i = 0; i < 8; ++i)
          kk[kbase + (long long)(mbase + i) * HD] = (__bf16)(acc[i] + bn);
      } else if (mt > 6 && mt < 22) {
        // interior q-tile: all 16 rows inside [100,356) -> unconditional
#pragma unroll
        for (int i = 0; i < 8; ++i)
          qc[qbase + (long long)(mbase + i) * HD] = (__bf16)(acc[i] + bn);
      } else {
        // partial q-tile (mt==6 or mt==22)
#pragma unroll
        for (int i = 0; i < 8; ++i) {
          const int m = mbase + i;
          if (m >= CSTART && m < CSTART + CSZ)
            qc[qbase + (long long)m * HD] = (__bf16)(acc[i] + bn);
        }
      }
    }
  }
}

// ---------------------------------------------------------------------------
// Kernel 2: flash attention per (window, head).  One workgroup per (w,h).
//   LDS: Q[256][32] + K[576][32] + V[32][576] (transposed) + P[8][16][32]
//   Q/K/V filled with async global->LDS b128 loads (ASYNCcnt path).
//   All WMMA fragment loads are contiguous 2x ds_load_b128.
// ---------------------------------------------------------------------------
__global__ __launch_bounds__(256) void halo_attn_kernel(
    const __bf16* __restrict__ qc, const __bf16* __restrict__ kk,
    const __bf16* __restrict__ vv, __bf16* __restrict__ aout) {
  extern __shared__ char smem[];
  __bf16* Q = (__bf16*)smem;        // 256*32
  __bf16* K = Q + CSZ * HD;         // 576*32   [key][d]
  __bf16* V = K + NTOK * HD;        // 32*576   [d][key]  (transposed)
  __bf16* P = V + NTOK * HD;        // 8*16*32

  const int wh = blockIdx.x;        // window*HEADS + head
  const int w  = wh >> 2, h = wh & 3;
  const int tid = threadIdx.x, lane = tid & 31, wv = tid >> 5;

  async_copy16(qc + (size_t)wh * CSZ  * HD, Q, CSZ  * HD * 2, tid);
  async_copy16(kk + (size_t)wh * NTOK * HD, K, NTOK * HD * 2, tid);
  async_copy16(vv + (size_t)wh * NTOK * HD, V, NTOK * HD * 2, tid);
  wait_async_all();
  __syncthreads();

  __bf16* myP = P + wv * 16 * HD;

  for (int t = 0; t < 2; ++t) {
    const int mt = wv * 2 + t;                       // 16 query tiles total
    v16bf aq = load_a16(&Q[(mt * 16) * HD], HD);     // K-dim = HD = 32: one frag
    v8f o0 = {}, o1 = {};
    float mrow[8], lrow[8];
#pragma unroll
    for (int i = 0; i < 8; ++i) { mrow[i] = -1e30f; lrow[i] = 0.0f; }

    for (int j = 0; j < NTOK / 32; ++j) {            // 18 chunks of 32 keys
      v16bf bk0 = load_bt16(&K[(j * 32 + 0)  * HD], HD);
      v16bf bk1 = load_bt16(&K[(j * 32 + 16) * HD], HD);
      v8f z = {};
      v8f s0 = wmma_bf16(aq, bk0, z);
      v8f s1 = wmma_bf16(aq, bk1, z);
#pragma unroll
      for (int i = 0; i < 8; ++i) {
        float a0 = s0[i] * SCALE, a1 = s1[i] * SCALE;
        float rm = fmaxf(a0, a1);                    // reduce over 16-lane half
#pragma unroll
        for (int off = 8; off >= 1; off >>= 1) rm = fmaxf(rm, __shfl_xor(rm, off, 32));
        const float mn   = fmaxf(mrow[i], rm);
        const float corr = __expf(mrow[i] - mn);
        const float p0 = __expf(a0 - mn), p1 = __expf(a1 - mn);
        float rs = p0 + p1;
#pragma unroll
        for (int off = 8; off >= 1; off >>= 1) rs += __shfl_xor(rs, off, 32);
        lrow[i] = lrow[i] * corr + rs;
        mrow[i] = mn;
        o0[i] *= corr; o1[i] *= corr;
        const int row = i + ((lane >> 4) << 3);      // C-layout row of this half
        myP[row * HD + (lane & 15)]      = (__bf16)p0;
        myP[row * HD + 16 + (lane & 15)] = (__bf16)p1;
      }
      // intra-wave LDS RAW (cross-lane re-layout of P): explicit wait
      asm volatile("s_wait_dscnt 0" ::: "memory");
      v16bf ap  = load_a16(myP, HD);
      // V transposed [d][key]: [N][K] storage -> contiguous b128 fragment loads
      v16bf bv0 = load_bt16(&V[j * 32], NTOK);              // d = 0..15
      v16bf bv1 = load_bt16(&V[16 * NTOK + j * 32], NTOK);  // d = 16..31
      o0 = wmma_bf16(ap, bv0, o0);
      o1 = wmma_bf16(ap, bv1, o1);
    }
#pragma unroll
    for (int i = 0; i < 8; ++i) {
      const float inv = 1.0f / lrow[i];
      const int row = i + ((lane >> 4) << 3);
      const int qi  = mt * 16 + row;
      const size_t base = ((size_t)w * CSZ + qi) * CC + h * HD;
      aout[base + (lane & 15)]      = (__bf16)(o0[i] * inv);
      aout[base + 16 + (lane & 15)] = (__bf16)(o1[i] * inv);
    }
  }
}

// ---------------------------------------------------------------------------
// Kernel 3: output projection + window-reverse.  One workgroup per window.
//   LDS: Y[256][128] bf16 (async fill) + Wp[128][128] bf16 + bias (~97 KB)
// ---------------------------------------------------------------------------
__global__ __launch_bounds__(256) void halo_proj_kernel(
    const __bf16* __restrict__ aout, const float* __restrict__ proj_w,
    const float* __restrict__ proj_b, float* __restrict__ out) {
  extern __shared__ char smem[];
  __bf16* Yl   = (__bf16*)smem;               // 256*128
  __bf16* Wp   = Yl + CSZ * CC;               // 128*128
  float*  bias = (float*)(Wp + CC * CC);      // 128

  const int w = blockIdx.x, tid = threadIdx.x, lane = tid & 31, wv = tid >> 5;

  async_copy16(aout + (size_t)w * CSZ * CC, Yl, CSZ * CC * 2, tid);
  for (int i = tid * 4; i < CC * CC; i += 1024) {
    float4 f = *(const float4*)&proj_w[i];
    v4bf b; b[0] = (__bf16)f.x; b[1] = (__bf16)f.y; b[2] = (__bf16)f.z; b[3] = (__bf16)f.w;
    *(v4bf*)&Wp[i] = b;
  }
  for (int i = tid; i < CC; i += 256) bias[i] = proj_b[i];
  wait_async_all();
  __syncthreads();

  for (int tile = wv; tile < 16 * 8; tile += 8) {
    const int mt = tile >> 3, nt = tile & 7;
    v8f acc = {};
#pragma unroll
    for (int k4 = 0; k4 < 4; ++k4) {
      v16bf a = load_a16(&Yl[(mt * 16) * CC + k4 * 32], CC);
      v16bf b = load_bt16(&Wp[(nt * 16) * CC + k4 * 32], CC);
      acc = wmma_bf16(a, b, acc);
    }
    const int ch  = nt * 16 + (lane & 15);
    const float bn = bias[ch];
    const int tok = mt * 16 + ((lane >> 4) << 3);   // 8 consecutive tokens
    float* dst = out + ((size_t)w * CC + ch) * 256 + tok;
    float4 lo = make_float4(acc[0] + bn, acc[1] + bn, acc[2] + bn, acc[3] + bn);
    float4 hi = make_float4(acc[4] + bn, acc[5] + bn, acc[6] + bn, acc[7] + bn);
    *(float4*)(dst)     = lo;
    *(float4*)(dst + 4) = hi;
  }
}

// ---------------------------------------------------------------------------

extern "C" void kernel_launch(void* const* d_in, const int* in_sizes, int n_in,
                              void* d_out, int out_size, void* d_ws, size_t ws_size,
                              hipStream_t stream) {
  const float* x      = (const float*)d_in[0];
  const float* qkv_w  = (const float*)d_in[1];
  const float* qkv_b  = (const float*)d_in[2];
  const float* proj_w = (const float*)d_in[3];
  const float* proj_b = (const float*)d_in[4];

  const int B = 512;
  char* ws = (char*)d_ws;
  size_t off = 0;
  __bf16* qc = (__bf16*)(ws + off); off += (size_t)B * HEADS * CSZ  * HD * 2; // 32 MB
  __bf16* kk = (__bf16*)(ws + off); off += (size_t)B * HEADS * NTOK * HD * 2; // 72 MB
  __bf16* vv = (__bf16*)(ws + off); off += (size_t)B * HEADS * NTOK * HD * 2; // 72 MB (transposed)
  __bf16* ao = (__bf16*)(ws + off); off += (size_t)B * CSZ * CC * 2;          // 32 MB
  (void)ws_size; (void)in_sizes; (void)n_in; (void)out_size;

  const size_t lds1 = (size_t)NTOK * CC * 2 + 3u * CC * CC * 2 + 3u * CC * 4; // ~242 KB
  const size_t lds2 = (size_t)CSZ * HD * 2 + 2u * NTOK * HD * 2 + 8u * 16 * HD * 2; // 96 KB
  const size_t lds3 = (size_t)CSZ * CC * 2 + (size_t)CC * CC * 2 + CC * 4;    // ~96.5 KB

  halo_qkv_kernel<<<B, 256, lds1, stream>>>(x, qkv_w, qkv_b, qc, kk, vv);
  halo_attn_kernel<<<B * HEADS, 256, lds2, stream>>>(qc, kk, vv, ao);
  halo_proj_kernel<<<B, 256, lds3, stream>>>(ao, proj_w, proj_b, (float*)d_out);
}
